// LSTMAutoencoder_46729244180983
// MI455X (gfx1250) — compile-verified, matrix operand order checked by reading, not run on hardware
//
#include <hip/hip_runtime.h>
#include <hip/hip_bf16.h>

typedef __attribute__((ext_vector_type(16))) __bf16 v16bf;
typedef __attribute__((ext_vector_type(8)))  float  v8f;

// Problem constants (match reference)
constexpr int Bc = 32, Fc = 128, Tc = 2400, Hc = 256, SEGc = 24, Sc = 100;

__device__ __forceinline__ v8f wmma_bf16(v16bf a, v16bf b, v8f c) {
  // D = A(16x32 bf16) x B(32x16 bf16) + C(16x16 f32)
  return __builtin_amdgcn_wmma_f32_16x16x32_bf16(
      /*neg_a=*/false, a, /*neg_b=*/false, b,
      /*c_mod=*/(short)0, c, /*reuse_a=*/false, /*reuse_b=*/false);
}

// Load a 16-element bf16 fragment slice for this lane: elements 0..7 at
// base[lk8 .. lk8+7], elements 8..15 at base[lk8+16 .. lk8+23].
// Matches ISA A/B 16-bit layout (K split across lane halves).
__device__ __forceinline__ v16bf load_frag16(const __bf16* base, int lk8) {
  union { uint4 u[2]; v16bf v; } r;
  r.u[0] = *reinterpret_cast<const uint4*>(base + lk8);
  r.u[1] = *reinterpret_cast<const uint4*>(base + lk8 + 16);
  return r.v;
}

__device__ __forceinline__ float sigm(float x) { return 1.0f / (1.0f + __expf(-x)); }

// MODE 0: outer encoder  — x gathered from inp (B,F,T) fp32; emit final c (bf16)
// MODE 1: inner encoder  — x from bf16 (B,S,H);              emit final c (bf16)
// MODE 2: inner decoder  — x constant = bottleneck (B,H);    emit h each step (bf16)
// MODE 3: outer decoder  — x constant = dec_out (B*S,H);     emit h each step (fp32, transposed)
template <int IN, int HID, int MODE, int TS>
__global__ __launch_bounds__(256) void lstm_scan(
    const float* __restrict__ xf, const __bf16* __restrict__ xbf,
    const __bf16* __restrict__ wih, const __bf16* __restrict__ whh,
    const float* __restrict__ bih, const float* __restrict__ bhh,
    __bf16* __restrict__ obf, float* __restrict__ of) {
  constexpr int TPW = (4 * HID / 16) / 8;  // gate tiles per wave (8 or 4)
  constexpr int NPG = TPW / 4;             // hidden-col tiles per wave (2 or 1)
  constexpr int KH  = HID / 32;            // recurrent K tiles
  constexpr int KI  = IN / 32;             // input K tiles

  __shared__ __align__(16) __bf16 hlds[16 * HID];

  const int tid  = threadIdx.x;
  const int wave = tid >> 5;
  const int lane = tid & 31;
  const int lm   = lane & 15;
  const int lk8  = (lane >> 4) * 8;
  const int rowbase = blockIdx.x * 16;

  // Accumulator init value: bias (and constant-input contribution for decoders)
  v8f base[TPW];
#pragma unroll
  for (int j = 0; j < TPW; ++j) {
    int col = (wave + 8 * j) * 16 + lm;   // global gate column for this lane
    float bv = bih[col] + bhh[col];
    v8f b;
#pragma unroll
    for (int r = 0; r < 8; ++r) b[r] = bv;
    base[j] = b;
  }

  if constexpr (MODE == 2 || MODE == 3) {
    // Input is time-invariant: fold x @ Wih^T into base once.
    const __bf16* xrow = xbf + (size_t)(rowbase + lm) * IN;
#pragma unroll
    for (int kt = 0; kt < KI; ++kt) {
      v16bf a = load_frag16(xrow + kt * 32, lk8);
#pragma unroll
      for (int j = 0; j < TPW; ++j) {
        const __bf16* wb = wih + (size_t)((wave + 8 * j) * 16 + lm) * IN + kt * 32;
        base[j] = wmma_bf16(a, load_frag16(wb, lk8), base[j]);
      }
    }
  }

  v8f creg[NPG];
#pragma unroll
  for (int q = 0; q < NPG; ++q) {
    v8f z = {0.f, 0.f, 0.f, 0.f, 0.f, 0.f, 0.f, 0.f};
    creg[q] = z;
  }

#pragma unroll 1
  for (int t = 0; t < TS; ++t) {
    v8f acc[TPW];
#pragma unroll
    for (int j = 0; j < TPW; ++j) acc[j] = base[j];

    if constexpr (MODE == 0) {
      // Gather A tile from inp (B,F,T): element (m=lm, k) at inp[b, k, s*SEG+t]
      const int row = rowbase + lm;
      const int bb = row / Sc, ss = row - bb * Sc;
      const float* xp = xf + (size_t)bb * Fc * Tc + (size_t)ss * SEGc + t;
#pragma unroll
      for (int kt = 0; kt < KI; ++kt) {
        v16bf a;
#pragma unroll
        for (int e = 0; e < 16; ++e) {
          int k = kt * 32 + lk8 + (e < 8 ? e : e + 8);
          a[e] = (__bf16)xp[(size_t)k * Tc];
        }
#pragma unroll
        for (int j = 0; j < TPW; ++j) {
          const __bf16* wb = wih + (size_t)((wave + 8 * j) * 16 + lm) * IN + kt * 32;
          acc[j] = wmma_bf16(a, load_frag16(wb, lk8), acc[j]);
        }
      }
    } else if constexpr (MODE == 1) {
      const __bf16* xrow = xbf + ((size_t)(rowbase + lm) * Sc + t) * IN;
#pragma unroll
      for (int kt = 0; kt < KI; ++kt) {
        v16bf a = load_frag16(xrow + kt * 32, lk8);
#pragma unroll
        for (int j = 0; j < TPW; ++j) {
          const __bf16* wb = wih + (size_t)((wave + 8 * j) * 16 + lm) * IN + kt * 32;
          acc[j] = wmma_bf16(a, load_frag16(wb, lk8), acc[j]);
        }
      }
    }

    if (t > 0) {  // h(0) == 0 -> recurrent term vanishes at t==0
#pragma unroll
      for (int kt = 0; kt < KH; ++kt) {
        v16bf a = load_frag16(&hlds[lm * HID + kt * 32], lk8);
#pragma unroll
        for (int j = 0; j < TPW; ++j) {
          const __bf16* wb = whh + (size_t)((wave + 8 * j) * 16 + lm) * HID + kt * 32;
          acc[j] = wmma_bf16(a, load_frag16(wb, lk8), acc[j]);
        }
      }
    }
    __syncthreads();  // all reads of hlds done before rewrite

    // Gate nonlinearity + state update. Stride-8 tile assignment guarantees
    // i/f/g/o for the wave's hidden columns are all in this wave's accumulators.
#pragma unroll
    for (int q = 0; q < NPG; ++q) {
      v8f iv = acc[0 * NPG + q];
      v8f fv = acc[1 * NPG + q];
      v8f gv = acc[2 * NPG + q];
      v8f ov = acc[3 * NPG + q];
      const int col = (wave + 8 * q) * 16 + lm;  // hidden column
#pragma unroll
      for (int r = 0; r < 8; ++r) {
        float cc = sigm(fv[r]) * creg[q][r] + sigm(iv[r]) * tanhf(gv[r]);
        creg[q][r] = cc;
        float hh = sigm(ov[r]) * tanhf(cc);
        int m = r + lk8;  // C/D layout: row = r + 8*(lane>=16)
        hlds[m * HID + col] = (__bf16)hh;
        if constexpr (MODE == 2) {
          obf[((size_t)(rowbase + m) * Sc + t) * HID + col] = (__bf16)hh;
        } else if constexpr (MODE == 3) {
          int row = rowbase + m;
          int bb = row / Sc, ss = row - bb * Sc;
          of[(size_t)bb * Fc * Tc + (size_t)col * Tc + ss * SEGc + t] = hh;
        }
      }
    }
    __syncthreads();  // h visible to all waves for next step
  }

  if constexpr (MODE == 0 || MODE == 1) {
#pragma unroll
    for (int q = 0; q < NPG; ++q) {
      const int col = (wave + 8 * q) * 16 + lm;
#pragma unroll
      for (int r = 0; r < 8; ++r) {
        int m = r + lk8;
        obf[(size_t)(rowbase + m) * HID + col] = (__bf16)creg[q][r];
      }
    }
  }
}

__global__ __launch_bounds__(256) void f2bf_kernel(const float* __restrict__ s,
                                                   __bf16* __restrict__ d, int n) {
  int i = blockIdx.x * 256 + threadIdx.x;
  if (i < n) d[i] = (__bf16)s[i];
}

extern "C" void kernel_launch(void* const* d_in, const int* in_sizes, int n_in,
                              void* d_out, int out_size, void* d_ws, size_t ws_size,
                              hipStream_t stream) {
  (void)in_sizes; (void)n_in; (void)out_size; (void)ws_size;
  const float* inp    = (const float*)d_in[0];
  const float* oe_wih = (const float*)d_in[1];
  const float* oe_whh = (const float*)d_in[2];
  const float* oe_bih = (const float*)d_in[3];
  const float* oe_bhh = (const float*)d_in[4];
  const float* ie_wih = (const float*)d_in[5];
  const float* ie_whh = (const float*)d_in[6];
  const float* ie_bih = (const float*)d_in[7];
  const float* ie_bhh = (const float*)d_in[8];
  const float* id_wih = (const float*)d_in[9];
  const float* id_whh = (const float*)d_in[10];
  const float* id_bih = (const float*)d_in[11];
  const float* id_bhh = (const float*)d_in[12];
  const float* od_wih = (const float*)d_in[13];
  const float* od_whh = (const float*)d_in[14];
  const float* od_bih = (const float*)d_in[15];
  const float* od_bhh = (const float*)d_in[16];
  float* out = (float*)d_out;

  char* ws = (char*)d_ws;
  size_t off = 0;
  auto take = [&](size_t nelem) -> __bf16* {
    __bf16* p = (__bf16*)(ws + off);
    off += (nelem * sizeof(__bf16) + 255) & ~(size_t)255;
    return p;
  };
  __bf16* oeW = take(1024 * 128);   // oe_wih bf16
  __bf16* oeU = take(1024 * 256);   // oe_whh bf16
  __bf16* ieW = take(1024 * 256);
  __bf16* ieU = take(1024 * 256);
  __bf16* idW = take(1024 * 256);
  __bf16* idU = take(1024 * 256);
  __bf16* odW = take(512 * 256);
  __bf16* odU = take(512 * 128);
  __bf16* cell = take((size_t)Bc * Sc * Hc);  // outer-encoder final c per segment
  __bf16* bott = take((size_t)Bc * Hc);       // inner-encoder bottleneck c
  __bf16* dec  = take((size_t)Bc * Sc * Hc);  // inner-decoder h sequence

  auto conv = [&](const float* s, __bf16* d, int n) {
    f2bf_kernel<<<(n + 255) / 256, 256, 0, stream>>>(s, d, n);
  };
  conv(oe_wih, oeW, 1024 * 128);
  conv(oe_whh, oeU, 1024 * 256);
  conv(ie_wih, ieW, 1024 * 256);
  conv(ie_whh, ieU, 1024 * 256);
  conv(id_wih, idW, 1024 * 256);
  conv(id_whh, idU, 1024 * 256);
  conv(od_wih, odW, 512 * 256);
  conv(od_whh, odU, 512 * 128);

  // 1) outer encoder: 3200 sequences x 24 steps -> cell (B,S,H)
  lstm_scan<128, 256, 0, SEGc><<<(Bc * Sc) / 16, 256, 0, stream>>>(
      inp, nullptr, oeW, oeU, oe_bih, oe_bhh, cell, nullptr);
  // 2) inner encoder: 32 sequences x 100 steps -> bottleneck (B,H)
  lstm_scan<256, 256, 1, Sc><<<Bc / 16, 256, 0, stream>>>(
      nullptr, cell, ieW, ieU, ie_bih, ie_bhh, bott, nullptr);
  // 3) inner decoder: 32 sequences x 100 steps (constant input) -> dec (B,S,H)
  lstm_scan<256, 256, 2, Sc><<<Bc / 16, 256, 0, stream>>>(
      nullptr, bott, idW, idU, id_bih, id_bhh, dec, nullptr);
  // 4) outer decoder: 3200 sequences x 24 steps (constant input) -> out (B,F,T)
  lstm_scan<256, 128, 3, SEGc><<<(Bc * Sc) / 16, 256, 0, stream>>>(
      nullptr, dec, odW, odU, od_bih, od_bhh, nullptr, out);
}
